// RGCLayer_41205916237988
// MI455X (gfx1250) — matrix operand order repaired and checked by reading, not compiled, not run on hardware
//
#include <hip/hip_runtime.h>

// ---------------------------------------------------------------------------
// R-GCN layer for MI455X (gfx1250, wave32).
//
// Phase A: supports[(dst*R+rel), d] += edge_val * vertex[src, d]   (bf16 accum
//          via GLOBAL_ATOMIC_PK_ADD_BF16, wave-per-edge, 2 feats/lane)
// Phase B: out = supports[N,1024] @ V[1024,128] + bias, with
//          V_WMMA_F32_16X16X32_BF16 (fp32 accumulate), V basis-combined and
//          pre-packed into the WMMA B-operand lane/VGPR layout, staged into
//          LDS with GLOBAL_LOAD_ASYNC_TO_LDS_B128.
//
// Constants from the reference: D=64, R=16, NB=8, H=128. N, E derived at launch.
// ---------------------------------------------------------------------------

#define Dq   64
#define Rq   16
#define NBq  8
#define Hq   128
#define KTOT 1024            // R*D (GEMM K)
#define NCHUNK 32            // KTOT / 32
#define NTILES 8             // H / 16

typedef __attribute__((ext_vector_type(16))) __bf16 v16bf;
typedef __attribute__((ext_vector_type(8)))  float  v8f;

typedef union { uint4 u[2]; v16bf v; } ABreg;

__device__ __forceinline__ unsigned short f2bf(float f) {
    unsigned int u = __float_as_uint(f);
    unsigned int r = u + 0x7FFFu + ((u >> 16) & 1u);   // round-to-nearest-even
    return (unsigned short)(r >> 16);
}

// ---------------------------------------------------------------- zero (b128)
__global__ void zero_ws_kernel(uint4* __restrict__ p, size_t n4) {
    size_t i  = (size_t)blockIdx.x * blockDim.x + threadIdx.x;
    size_t st = (size_t)gridDim.x * blockDim.x;
    uint4 z = {0u, 0u, 0u, 0u};
    for (; i < n4; i += st) p[i] = z;
}

// ------------------------------------------------- basis contraction + B-pack
// V_flat[c,h] = sum_b W_comp[c%R, b] * W[b*D + c/R, h];   c = GEMM K index.
// Packed to this kernel's B-operand layout, contiguous k = 16*hi + elem as A uses (32x16 tile, K chunk of 32):
//   hi = (kk>=16), lane = n + 16*hi, vgpr = (kk&15)>>1, half = kk&1
// Flat bf16 index: ((((nt*32 + chunk)*32 + lane)*8 + v)*2 + half
__global__ void build_B_kernel(const float* __restrict__ W,
                               const float* __restrict__ Wc,
                               unsigned short* __restrict__ Bpk) {
    int idx = blockIdx.x * blockDim.x + threadIdx.x;   // 0 .. KTOT*H-1
    if (idx >= KTOT * Hq) return;
    int k = idx >> 7;          // 0..1023
    int h = idx & (Hq - 1);    // 0..127
    int d = k >> 4;            // c / R
    int r = k & (Rq - 1);      // c % R
    float s = 0.f;
#pragma unroll
    for (int b = 0; b < NBq; ++b)
        s += Wc[r * NBq + b] * W[(b * Dq + d) * Hq + h];

    int chunk = k >> 5, kk = k & 31;
    int nt = h >> 4,   n  = h & 15;
    int hi   = kk >> 4;                 // k>=16 within chunk -> lanes 16..31
    int lane = n + 16 * hi;
    int v    = (kk & 15) >> 1;
    int hw   = kk & 1;
    size_t pos = ((((size_t)nt * NCHUNK + chunk) * 32 + lane) * 8 + v) * 2 + hw;
    Bpk[pos] = f2bf(s);
}

// ----------------------------------------------------------- edge scatter-add
// One wave per edge. Lane l handles features 2l, 2l+1 (one packed bf16x2 dword)
// -> 8B/lane coalesced vertex read, one GLOBAL_ATOMIC_PK_ADD_BF16 per lane.
__global__ __launch_bounds__(256)
void scatter_kernel(const float* __restrict__ vertex,
                    const float* __restrict__ eval,
                    const int*   __restrict__ esrc,
                    const int*   __restrict__ edst,
                    const int*   __restrict__ erel,
                    unsigned short* __restrict__ supports,  // bf16 [N*R, D]
                    int E) {
    int wv   = (int)((blockIdx.x * blockDim.x + threadIdx.x) >> 5);
    int lane = threadIdx.x & 31;
    if (wv >= E) return;

    int   src = esrc[wv];
    int   dst = edst[wv];
    int   rel = erel[wv];
    float val = eval[wv];

    const float* vr = vertex + (size_t)src * Dq + lane * 2;
    float a = vr[0] * val;
    float b = vr[1] * val;
    unsigned int pk = ((unsigned int)f2bf(b) << 16) | (unsigned int)f2bf(a);

    unsigned long long addr = (unsigned long long)
        (supports + ((size_t)dst * Rq + rel) * Dq + lane * 2);
    asm volatile("global_atomic_pk_add_bf16 %0, %1, off scope:SCOPE_DEV"
                 :: "v"(addr), "v"(pk) : "memory");
}

// ----------------------------------------------------------------------- GEMM
// out[N,128] = supports[N,1024](bf16) @ V(bf16, pre-packed) + bias, fp32 acc.
// Block = 256 threads (8 waves). Whole packed V (256 KB) staged in LDS once
// via async-to-LDS DMA. Each wave: one 16-row slab x all 128 cols = 8 C tiles.
// Per K-chunk: preload ALL 8 B tile register sets (so the 8 WMMAs issue
// back-to-back with no per-WMMA ds waits / hazard NOPs), prefetch next A.
__global__ __launch_bounds__(256)
void gemm_kernel(const unsigned short* __restrict__ S,    // bf16 supports
                 const unsigned int*   __restrict__ Bpk,  // packed V, dwords
                 const float* __restrict__ bias,
                 float* __restrict__ out,
                 int N, int Mtiles) {
    extern __shared__ unsigned int lB[];                  // 65536 dwords = 256KB

    // async DMA of packed V into LDS (no VGPR round trip, ASYNCcnt-tracked)
    {
        const uint4* g = (const uint4*)Bpk;
        uint4*       s = (uint4*)lB;
        for (int i = threadIdx.x; i < (KTOT * Hq / 2) / 4; i += 256) {
            unsigned int       ldsOff = (unsigned int)(uintptr_t)(s + i);
            unsigned long long ga     = (unsigned long long)(g + i);
            asm volatile("global_load_async_to_lds_b128 %0, %1, off"
                         :: "v"(ldsOff), "v"(ga) : "memory");
        }
        asm volatile("s_wait_asynccnt 0x0" ::: "memory");
    }
    __syncthreads();

    int wave = threadIdx.x >> 5;
    int lane = threadIdx.x & 31;
    int hl   = lane >> 4;        // half-wave: selects K sub-pattern of A
    int l16  = lane & 15;

    for (int tile = blockIdx.x * 8 + wave; tile < Mtiles; tile += gridDim.x * 8) {
        int rowA  = tile * 16 + l16;
        int rowAc = rowA < N ? rowA : N - 1;              // tail clamp (loads)
        // A lane base: row rowAc, K offset 0 (lanes 0-15) or 16 (lanes 16-31)
        const unsigned short* arow = S + (size_t)rowAc * KTOT + hl * 16;

        v8f acc[NTILES];
#pragma unroll
        for (int nt = 0; nt < NTILES; ++nt) acc[nt] = v8f{};

        // software pipeline: A for chunk 0 in flight before the loop
        ABreg Acur, Anext;
        Acur.u[0] = *(const uint4*)(arow);
        Acur.u[1] = *(const uint4*)(arow + 8);

#pragma unroll 2
        for (int c = 0; c < NCHUNK; ++c) {
            // preload all 8 B tiles for this chunk into distinct registers
            ABreg Bt[NTILES];
#pragma unroll
            for (int nt = 0; nt < NTILES; ++nt) {
                const uint4* bp =
                    (const uint4*)(lB + (((nt * NCHUNK + c) * 32 + lane) * 8));
                Bt[nt].u[0] = bp[0];
                Bt[nt].u[1] = bp[1];
            }
            // prefetch next chunk's A while B ds_loads are in flight
            if (c + 1 < NCHUNK) {
                Anext.u[0] = *(const uint4*)(arow + (c + 1) * 32);
                Anext.u[1] = *(const uint4*)(arow + (c + 1) * 32 + 8);
            }
            // 8 back-to-back WMMAs, A registers reused across all N-tiles
#pragma unroll
            for (int nt = 0; nt < NTILES; ++nt)
                acc[nt] = __builtin_amdgcn_wmma_f32_16x16x32_bf16(
                    false, Acur.v, false, Bt[nt].v, (short)0, acc[nt],
                    false, false);
            Acur = Anext;
        }

        // C/D layout: VGPR j -> row j + 8*(lane>=16), col = lane%16
#pragma unroll
        for (int nt = 0; nt < NTILES; ++nt) {
            float bv = bias[nt * 16 + l16];
            int   rb = tile * 16 + hl * 8;
#pragma unroll
            for (int j = 0; j < 8; ++j) {
                int row = rb + j;
                if (row < N)
                    out[(size_t)row * Hq + nt * 16 + l16] = acc[nt][j] + bv;
            }
        }
    }
}

// ---------------------------------------------------------------------------
extern "C" void kernel_launch(void* const* d_in, const int* in_sizes, int n_in,
                              void* d_out, int out_size, void* d_ws, size_t ws_size,
                              hipStream_t stream) {
    const float* vertex = (const float*)d_in[0];
    const float* eval   = (const float*)d_in[1];
    const float* W      = (const float*)d_in[2];
    const float* Wc     = (const float*)d_in[3];
    const float* bias   = (const float*)d_in[4];
    const int*   esrc   = (const int*)d_in[5];
    const int*   edst   = (const int*)d_in[6];
    const int*   erel   = (const int*)d_in[7];

    int N = in_sizes[0] / Dq;     // nodes
    int E = in_sizes[1];          // edges

    // workspace layout: [ supports bf16 : N*R*D*2 B ][ packed V bf16 : 256 KB ]
    size_t supBytes = (size_t)N * Rq * Dq * sizeof(unsigned short);
    unsigned short* supports = (unsigned short*)d_ws;
    unsigned short* Bpk      = (unsigned short*)((char*)d_ws + supBytes);
    float*          out      = (float*)d_out;

    // 1) zero bf16 supports (grid-stride b128 stores)
    {
        size_t n4 = supBytes / 16;
        zero_ws_kernel<<<4096, 256, 0, stream>>>((uint4*)supports, n4);
    }
    // 2) basis contraction + WMMA B-layout pack (131072 elements)
    build_B_kernel<<<(KTOT * Hq + 255) / 256, 256, 0, stream>>>(W, Wc, Bpk);
    // 3) edge scatter, one wave (32 lanes) per edge
    {
        int blocks = (E + 7) / 8;    // 8 waves per 256-thread block
        scatter_kernel<<<blocks, 256, 0, stream>>>(vertex, eval, esrc, edst,
                                                   erel, supports, E);
    }
    // 4) WMMA GEMM + bias, 256 KB dynamic LDS for packed V
    {
        int Mtiles = (N + 15) / 16;
        int blocks = (Mtiles + 7) / 8;
        gemm_kernel<<<blocks, 256, (size_t)(KTOT * Hq) * sizeof(unsigned short),
                      stream>>>(supports, (const unsigned int*)Bpk, bias, out,
                                N, Mtiles);
    }
}
